// ClusterModel_55267639164929
// MI455X (gfx1250) — matrix operand order repaired and checked by reading, not compiled, not run on hardware
//
#include <hip/hip_runtime.h>
#include <hip/hip_bf16.h>
#include <stdint.h>

// ---------------------------------------------------------------------------
// Problem constants (match reference)
// ---------------------------------------------------------------------------
#define NTOT 131072   // nodes
#define DDIM 128      // input dim
#define KEXP 1024     // expanded dim
#define HDIM 512      // hidden dim
#define NGRP 4        // groups
#define GN   32768    // nodes per group
#define FSEG 8192     // fine clusters
#define GSEG 32       // coarse graphs
#define CCLS 16       // classes

#define HS   1048     // H-tile LDS row stride (2096B: 16B*odd -> conflict-free)
#define BSS  136      // B-stage LDS row stride (272B = 256B + 16B pad)
// LDS: 64*HS*2 (H tile, 134144B) + 512*BSS*2 (B stage, 139264B) = 273408B
#define SMEM_BYTES (64 * HS * 2 + 512 * BSS * 2)

typedef __bf16 bf16_t;
typedef __attribute__((ext_vector_type(16))) __bf16 v16bf;
typedef __attribute__((ext_vector_type(8)))  __bf16 v8bf;
typedef __attribute__((ext_vector_type(8)))  float  v8f;
typedef int v4i __attribute__((vector_size(16)));   // 16B payload for async LDS copy

typedef unsigned int u32x4 __attribute__((ext_vector_type(4)));
typedef int          i32x8 __attribute__((ext_vector_type(8)));
typedef int          i32x4 __attribute__((ext_vector_type(4)));

#if __has_builtin(__builtin_amdgcn_global_load_async_to_lds_b128)
#define USE_ASYNC 1
#endif
#if __has_builtin(__builtin_amdgcn_tensor_load_to_lds)
#define USE_TDM 1
#endif

// Build a v16bf fragment from two 16-byte chunks (ISA lane layout).
static __device__ inline v16bf load2x8(const bf16_t* p0, const bf16_t* p1) {
  v8bf lo = *(const v8bf*)p0;
  v8bf hi = *(const v8bf*)p1;
  return __builtin_shufflevector(lo, hi, 0,1,2,3,4,5,6,7,8,9,10,11,12,13,14,15);
}

// float atomic max via integer atomics (sign-aware; emits global_atomic_max/min)
static __device__ inline void atomicMaxF(float* addr, float val) {
  if (val >= 0.0f) atomicMax((int*)addr, __float_as_int(val));
  else             atomicMin((unsigned int*)addr, __float_as_uint(val));
}

// Stage a [512 rows][128 k] bf16 panel (row stride srcStride elems) from global
// into LDS at row stride BSS, via ASYNCcnt-tracked global->LDS copies.
static __device__ inline void stage512x128(bf16_t* __restrict__ dst,
                                           const bf16_t* __restrict__ src,
                                           int srcStride, int tid) {
#pragma unroll 4
  for (int i = 0; i < 32; ++i) {
    int c   = i * 256 + tid;
    int row = c >> 4;
    int ko  = (c & 15) * 8;                 // element offset (16B chunks)
    const bf16_t* s = src + (size_t)row * srcStride + ko;
    bf16_t*       d = dst + (size_t)row * BSS + ko;
#ifdef USE_ASYNC
    __builtin_amdgcn_global_load_async_to_lds_b128(
        (__attribute__((address_space(1))) v4i*)(s),
        (__attribute__((address_space(3))) v4i*)(d), 0, 0);
#else
    *(uint4*)d = *(const uint4*)s;
#endif
  }
}

static __device__ inline void asyncWait0() {
#ifdef USE_ASYNC
#if __has_builtin(__builtin_amdgcn_s_wait_asynccnt)
  __builtin_amdgcn_s_wait_asynccnt(0);
#else
  asm volatile("s_wait_asynccnt 0x0" ::: "memory");
#endif
#endif
}

#ifdef USE_TDM
// Tensor Data Mover: one instruction stages a [512 rows][128 k] bf16 panel
// into LDS, with hardware row padding (256B data + 16B pad = BSS stride).
// D# packing per CDNA5 ISA ch.8 (group0: count/lds_addr/global_addr/type;
// group1: data_size=2B, pad 16B per 256B, tensor 128x512, tile 128x512).
// This toolchain's builtin takes 6 args (g0, g1, g2, g3, extra i32x8, cpol).
static __device__ inline void tdm_stage(unsigned ldsByteOff,
                                        const bf16_t* src, int strideElems) {
  unsigned long long ga = (unsigned long long)(uintptr_t)src;
  u32x4 g0;
  g0[0] = 1u;                                           // count=1 (valid D#)
  g0[1] = ldsByteOff;                                   // lds_addr (bytes)
  g0[2] = (unsigned)ga;                                 // global_addr[31:0]
  g0[3] = (unsigned)((ga >> 32) & 0x1FFFFFFu) | (2u << 30);  // addr[56:32]|type=2
  i32x8 g1;
  g1[0] = (1 << 16)            // data_size = 1 -> 2 bytes
        | (1 << 20)            // pad_enable
        | (5 << 22)            // pad_interval: 64 DWORDs = 256B
        | (3 << 25);           // pad_amount: 4 DWORDs = 16B
  g1[1] = (128 << 16);         // tensor_dim0 = 128 elems (bits 79:48, low half)
  g1[2] = (512 << 16);         // tensor_dim0 hi | tensor_dim1 = 512 (low half)
  g1[3] = (128 << 16);         // tensor_dim1 hi | tile_dim0 = 128
  g1[4] = 512;                 // tile_dim1 = 512, tile_dim2 = 0
  g1[5] = strideElems;         // tensor_dim0_stride[31:0] (elems)
  g1[6] = 0;                   // dim0_stride hi | dim1_stride lo (2D: unused)
  g1[7] = 0;
  i32x4 z4 = {0, 0, 0, 0};                 // groups 2/3: unused (2D tensor)
  i32x8 z8 = {0, 0, 0, 0, 0, 0, 0, 0};     // extra descriptor group: unused
  __builtin_amdgcn_tensor_load_to_lds(g0, g1, z4, z4, z8, 0);
}

static __device__ inline void tensorWait0() {
#if __has_builtin(__builtin_amdgcn_s_wait_tensorcnt)
  __builtin_amdgcn_s_wait_tensorcnt(0);
#else
  asm volatile("s_wait_tensorcnt 0x0" ::: "memory");
#endif
}
#endif  // USE_TDM

// ---------------------------------------------------------------------------
// Prep kernels
// ---------------------------------------------------------------------------
__global__ void init_emb_kernel(uint32_t* __restrict__ emb) {
  int o = blockIdx.x * blockDim.x + threadIdx.x;
  if (o < FSEG * HDIM) emb[o] = 0xFF800000u;   // -inf
}

__global__ void gather_convert_x_kernel(const float* __restrict__ x,
                                        const int* __restrict__ gidx,
                                        bf16_t* __restrict__ xg) {
  int o = blockIdx.x * blockDim.x + threadIdx.x;     // over NTOT*DDIM
  int d  = o & (DDIM - 1);
  int gn = o >> 7;
  int node = gidx[gn];
  xg[o] = (bf16_t)x[(size_t)node * DDIM + d];
}

__global__ void transpose_w1_kernel(const float* __restrict__ W1,
                                    bf16_t* __restrict__ W1t) {
  int o = blockIdx.x * blockDim.x + threadIdx.x;     // over NGRP*KEXP*DDIM
  int d = o & (DDIM - 1);
  int r = o >> 7;
  int n = r & (KEXP - 1);
  int g = r >> 10;
  W1t[o] = (bf16_t)W1[((size_t)g * DDIM + d) * KEXP + n];
}

__global__ void transpose_w2_kernel(const float* __restrict__ W2,
                                    bf16_t* __restrict__ W2t) {
  int o = blockIdx.x * blockDim.x + threadIdx.x;     // over NGRP*HDIM*KEXP
  int k = o & (KEXP - 1);
  int r = o >> 10;
  int h = r & (HDIM - 1);
  int g = r >> 9;
  W2t[o] = (bf16_t)W2[((size_t)g * KEXP + k) * HDIM + h];
}

// ---------------------------------------------------------------------------
// Fused: GEMM1(relu) -> LDS -> GEMM2 -> atomic segment-max into emb
// grid = NGRP * (GN/64) = 2048 blocks, 256 threads (8 waves)
// B panels staged to LDS once per block: GEMM1 via async global->LDS,
// GEMM2 via Tensor Data Mover (TDM).
// ---------------------------------------------------------------------------
__global__ __launch_bounds__(256)
void fused_mlp_kernel(const bf16_t* __restrict__ xg,    // [NGRP][GN][128]
                      const bf16_t* __restrict__ W1t,   // [NGRP][1024][128]
                      const bf16_t* __restrict__ W2t,   // [NGRP][512][1024]
                      const float*  __restrict__ b1,    // [NGRP][1024]
                      const float*  __restrict__ b2,    // [NGRP][512]
                      const int*    __restrict__ gidx,  // [NGRP][GN]
                      const int*    __restrict__ fine,  // [NTOT]
                      float*        __restrict__ emb)   // [FSEG][512]
{
  extern __shared__ char smem[];
  bf16_t* hTile = (bf16_t*)smem;                 // [64][HS]
  bf16_t* Bs    = (bf16_t*)(smem + 64 * HS * 2); // [512][BSS]

  const int tid  = threadIdx.x;
  const int wave = tid >> 5;
  const int lane = tid & 31;
  const int hf   = lane >> 4;        // lane half (ISA K-striping)
  const int l16  = lane & 15;

  const int g       = blockIdx.x >> 9;      // 512 tiles per group
  const int tile    = blockIdx.x & 511;
  const int rowBase = tile * 64;
  const int mrow    = (wave & 3) * 16;      // 4 row-waves
  const int nhalf   = wave >> 2;            // 2 col-waves

  // ---------------- GEMM1: [16 rows] x [512 cols], K = 128 ----------------
  // A fragments (16x32 bf16) cached in VGPRs for the whole phase.
  const bf16_t* xrow = xg + ((size_t)g * GN + rowBase + mrow + l16) * DDIM;
  v16bf afr1[4];
#pragma unroll
  for (int kb = 0; kb < 4; ++kb) {
    int off = kb * 32 + hf * 8;            // lanes<16: K 0..7,16..23 ; >=16: +8
    afr1[kb] = load2x8(xrow + off, xrow + off + 16);
  }

  const bf16_t* w1g = W1t + (size_t)g * KEXP * DDIM;
  const float*  b1g = b1 + g * KEXP;

  for (int p = 0; p < 2; ++p) {            // two 512-col panels of W1t
    __syncthreads();                       // previous Bs consumers done
    stage512x128(Bs, w1g + (size_t)p * 512 * DDIM, DDIM, tid);
    asyncWait0();
    __syncthreads();

#pragma unroll 2
    for (int nb = 0; nb < 16; ++nb) {
      const int nloc = nhalf * 256 + nb * 16;          // row in Bs
      const bf16_t* brow = Bs + (size_t)(nloc + l16) * BSS + hf * 8;
      v8f acc = {};
#pragma unroll
      for (int kb = 0; kb < 4; ++kb) {
        v16bf bfr = load2x8(brow + kb * 32, brow + kb * 32 + 16);
        acc = __builtin_amdgcn_wmma_f32_16x16x32_bf16(
            false, afr1[kb], false, bfr, (short)0, acc, false, false);
      }
      const float bias = b1g[p * 512 + nloc + l16];
      bf16_t* hdst = hTile + (size_t)(mrow + hf * 8) * HS + p * 512 + nloc + l16;
#pragma unroll
      for (int r = 0; r < 8; ++r) {                    // D elem r -> row +r
        float v = acc[r] + bias;
        v = v > 0.0f ? v : 0.0f;                       // ReLU
        hdst[(size_t)r * HS] = (bf16_t)v;
      }
    }
  }

  // -------------- GEMM2: [16 rows] x [256 cols], K = 1024 ------------------
  const bf16_t* w2g  = W2t + (size_t)g * HDIM * KEXP;
  const float*  b2g  = b2 + g * HDIM;
  const bf16_t* hrow = hTile + (size_t)(mrow + l16) * HS;

#ifdef USE_TDM
  const unsigned bsOff =
      (unsigned)(unsigned long long)(__attribute__((address_space(3))) char*)(void*)Bs;
#endif

  // Scatter targets: fine cluster id per output row of this wave.
  const int rbaseG = rowBase + mrow + hf * 8;
  int fidx[8];
#pragma unroll
  for (int r = 0; r < 8; ++r) {
    int node = gidx[g * GN + rbaseG + r];
    fidx[r] = fine[node];
  }

  v8f acc[16] = {};                        // 16 rows x 256 cols per wave

  for (int kc = 0; kc < 8; ++kc) {         // K chunks of 128
    __syncthreads();                       // previous Bs consumers done
#ifdef USE_TDM
    if (wave == 0) {                       // one TDM op moves the whole panel
      tdm_stage(bsOff, w2g + (size_t)kc * 128, KEXP);
      tensorWait0();
    }
    __syncthreads();
#else
    stage512x128(Bs, w2g + (size_t)kc * 128, KEXP, tid);   // [512 h][128 k]
    asyncWait0();
    __syncthreads();
#endif

#pragma unroll
    for (int kb = 0; kb < 4; ++kb) {
      const int koff = kc * 128 + kb * 32 + hf * 8;
      v16bf afr = load2x8(hrow + koff, hrow + koff + 16);  // A from H tile
#pragma unroll
      for (int nb = 0; nb < 16; ++nb) {
        const bf16_t* brow =
            Bs + (size_t)(nhalf * 256 + nb * 16 + l16) * BSS + kb * 32 + hf * 8;
        v16bf bfr = load2x8(brow, brow + 16);
        acc[nb] = __builtin_amdgcn_wmma_f32_16x16x32_bf16(
            false, afr, false, bfr, (short)0, acc[nb], false, false);
      }
    }
  }

  // Epilogue: bias + fused segment-max scatter.
#pragma unroll 2
  for (int nb = 0; nb < 16; ++nb) {
    const int   col  = nhalf * 256 + nb * 16 + l16;
    const float bias = b2g[col];
#pragma unroll
    for (int r = 0; r < 8; ++r) {
      atomicMaxF(emb + (size_t)fidx[r] * HDIM + col, acc[nb][r] + bias);
    }
  }
}

// ---------------------------------------------------------------------------
// InstanceNorm stats: per (coarse graph, channel) mean & rstd over 256 rows
// ---------------------------------------------------------------------------
__global__ void norm_stats_kernel(const float* __restrict__ emb,
                                  float* __restrict__ meanA,
                                  float* __restrict__ rstdA) {
  const int gph = blockIdx.x;                         // 32 graphs
  for (int c = threadIdx.x; c < HDIM; c += blockDim.x) {
    const float* base = emb + ((size_t)gph * 256) * HDIM + c;
    float s = 0.0f, s2 = 0.0f;
    for (int r = 0; r < 256; ++r) {
      float v = base[(size_t)r * HDIM];
      s += v; s2 += v * v;
    }
    float mean = s * (1.0f / 256.0f);
    float var  = s2 * (1.0f / 256.0f) - mean * mean;
    meanA[gph * HDIM + c] = mean;
    rstdA[gph * HDIM + c] = rsqrtf(var + 1e-5f);
  }
}

// ---------------------------------------------------------------------------
// Classifier: logits[f][c] = sum_h normed(f,h) * w_out[h][c] + b_out[c]
// ---------------------------------------------------------------------------
__global__ void logits_kernel(const float* __restrict__ emb,
                              const float* __restrict__ meanA,
                              const float* __restrict__ rstdA,
                              const int*   __restrict__ bcc,    // [FSEG]
                              const float* __restrict__ w_out,  // [512][16]
                              const float* __restrict__ b_out,  // [16]
                              float*       __restrict__ out) {  // [FSEG][16]
  const int o = blockIdx.x * blockDim.x + threadIdx.x;          // FSEG*CCLS
  const int c = o & (CCLS - 1);
  const int f = o >> 4;
  const int gph = bcc[f];
  const float* er = emb   + (size_t)f * HDIM;
  const float* mr = meanA + gph * HDIM;
  const float* rr = rstdA + gph * HDIM;
  float acc = b_out[c];
#pragma unroll 4
  for (int h = 0; h < HDIM; ++h) {
    float v = (er[h] - mr[h]) * rr[h];
    acc += v * w_out[h * CCLS + c];
  }
  out[o] = acc;
}

// ---------------------------------------------------------------------------
// Launch
// ---------------------------------------------------------------------------
extern "C" void kernel_launch(void* const* d_in, const int* in_sizes, int n_in,
                              void* d_out, int out_size, void* d_ws, size_t ws_size,
                              hipStream_t stream) {
  const float* x     = (const float*)d_in[0];
  const int*   gidx  = (const int*)  d_in[1];   // [NGRP*GN]
  const int*   fine  = (const int*)  d_in[2];   // [NTOT]
  const int*   bcc   = (const int*)  d_in[3];   // [FSEG]
  const float* W1    = (const float*)d_in[4];
  const float* b1    = (const float*)d_in[5];
  const float* W2    = (const float*)d_in[6];
  const float* b2    = (const float*)d_in[7];
  const float* w_out = (const float*)d_in[8];
  const float* b_out = (const float*)d_in[9];
  float* out = (float*)d_out;

  // Workspace carve (all sizes are power-of-two multiples; 256B aligned)
  char* p = (char*)d_ws;
  bf16_t* xg   = (bf16_t*)p; p += (size_t)NTOT * DDIM * 2;        // 32 MB
  bf16_t* W1t  = (bf16_t*)p; p += (size_t)NGRP * KEXP * DDIM * 2; //  1 MB
  bf16_t* W2t  = (bf16_t*)p; p += (size_t)NGRP * HDIM * KEXP * 2; //  4 MB
  float*  emb  = (float*)p;  p += (size_t)FSEG * HDIM * 4;        // 16 MB
  float*  meanA = (float*)p; p += (size_t)GSEG * HDIM * 4;
  float*  rstdA = (float*)p; p += (size_t)GSEG * HDIM * 4;

  // Prep
  init_emb_kernel<<<(FSEG * HDIM) / 256, 256, 0, stream>>>((uint32_t*)emb);
  gather_convert_x_kernel<<<(NTOT * DDIM) / 256, 256, 0, stream>>>(x, gidx, xg);
  transpose_w1_kernel<<<(NGRP * KEXP * DDIM) / 256, 256, 0, stream>>>(W1, W1t);
  transpose_w2_kernel<<<(NGRP * HDIM * KEXP) / 256, 256, 0, stream>>>(W2, W2t);

  // Fused MLP + segment-max (273,408 B dynamic LDS: CDNA5 WGP has 320 KB)
  fused_mlp_kernel<<<NGRP * (GN / 64), 256, SMEM_BYTES, stream>>>(
      xg, W1t, W2t, b1, b2, gidx, fine, emb);

  // InstanceNorm + classifier
  norm_stats_kernel<<<GSEG, 256, 0, stream>>>(emb, meanA, rstdA);
  logits_kernel<<<(FSEG * CCLS) / 256, 256, 0, stream>>>(
      emb, meanA, rstdA, bcc, w_out, b_out, out);
}
